// STHG_4810363373047
// MI455X (gfx1250) — matrix operand order repaired and checked by reading, not compiled, not run on hardware
//
#include <hip/hip_runtime.h>
#include <math.h>
#include <cstdint>

// ---------------------------------------------------------------------------
// CDNA5 (gfx1250) WMMA transformer pipeline, wave32.
// - all GEMM/attention math on v_wmma_f32_16x16x32_f16 (f32 accumulate)
// - weights pre-converted to f16 once per launch (L2-resident, halves bytes)
// - activation A-tiles staged to LDS with GLOBAL_LOAD_ASYNC_TO_LDS_B128
//   (ASYNCcnt + s_wait_asynccnt), since producers emit f16 copies directly
// ---------------------------------------------------------------------------

typedef __attribute__((ext_vector_type(16))) _Float16 v16h;
typedef __attribute__((ext_vector_type(8)))  _Float16 v8h;
typedef __attribute__((ext_vector_type(8)))  float    v8f;
typedef int v4i __attribute__((vector_size(16)));
typedef __attribute__((address_space(1))) v4i gv4i;  // global AS int4
typedef __attribute__((address_space(3))) v4i lv4i;  // LDS AS int4

#define TT 32
#define NN 512
#define DD 128
#define FFD 2048
#define HH 8
#define MTOK (TT * NN)  // 16384 tokens
#define WNDW 8

// ---------------- CDNA5 async global->LDS copy (16 bytes / lane) -----------
__device__ __forceinline__ void cp_async_b128(const _Float16* gsrc,
                                              _Float16* ldst) {
#if __has_builtin(__builtin_amdgcn_global_load_async_to_lds_b128)
  __builtin_amdgcn_global_load_async_to_lds_b128(
      (gv4i*)(v4i*)(void*)const_cast<_Float16*>(gsrc),
      (lv4i*)(v4i*)(void*)ldst, 0, 0);
#elif defined(__AMDGCN__)
  asm volatile("global_load_async_to_lds_b128 %0, %1, off"
               :
               : "v"((unsigned)(uintptr_t)ldst), "v"(gsrc)
               : "memory");
#else
  __builtin_memcpy(ldst, gsrc, 16);  // host semantic pass only
#endif
}

__device__ __forceinline__ void wait_async() {
#if __has_builtin(__builtin_amdgcn_s_wait_asynccnt)
  __builtin_amdgcn_s_wait_asynccnt(0);
#elif defined(__AMDGCN__)
  asm volatile("s_wait_asynccnt 0" ::: "memory");
#endif
}

// D = A(16x32 f16) * B(32x16 f16) + C(16x16 f32)
__device__ __forceinline__ v8f wmma16(v16h a, v16h b, v8f c) {
  return __builtin_amdgcn_wmma_f32_16x16x32_f16(false, a, false, b, (short)0, c,
                                                false, false);
}

// A-matrix fragment from LDS (row-major f16 tile, `stride` halfs).
// lanes 0-15: row=lane, K=k0+{0..15}; lanes 16-31: row=lane-16, K+=16.
__device__ __forceinline__ v16h lds_a_frag(const _Float16* base, int stride,
                                           int k0) {
  const int lane = threadIdx.x & 31;
  const int kb = k0 + ((lane >> 4) << 4);
  const _Float16* p = base + (lane & 15) * stride;
  v16h a;
#pragma unroll
  for (int i = 0; i < 8; ++i) a[i] = p[kb + i];
#pragma unroll
  for (int i = 0; i < 8; ++i) a[8 + i] = p[kb + 8 + i];
  return a;
}

// B-matrix fragment from row-major f16 weights W[Nout][ldw] (Y = X*W^T):
// lane column n = lane%16 (a W row), contiguous K = k0 + 16*(lane/16)..+15.
__device__ __forceinline__ v16h glb_b_frag(const _Float16* __restrict__ W,
                                           int ldw, int n0, int k0) {
  const int lane = threadIdx.x & 31;
  const _Float16* p =
      W + (long)(n0 + (lane & 15)) * ldw + k0 + ((lane >> 4) << 4);
  const v8h lo = *(const v8h*)p;
  const v8h hi = *(const v8h*)(p + 8);
  v16h b;
#pragma unroll
  for (int i = 0; i < 8; ++i) {
    b[i] = lo[i];
    b[8 + i] = hi[i];
  }
  return b;
}

// ---------------------------------------------------------------------------
// GEMM: Y[:, col_base+0..128) = X(f16)[M,K] @ W(f16)^T + bias (+ resid f32)
// ---------------------------------------------------------------------------
template <bool RESID, bool OUTH>
__global__ __launch_bounds__(128) void gemm_kernel(
    const _Float16* __restrict__ X, int ldx, const _Float16* __restrict__ W,
    int ldw, const float* __restrict__ bias, const float* __restrict__ resid,
    float* __restrict__ Yf, _Float16* __restrict__ Yh, int ldy, int col_base,
    int K) {
  __shared__ _Float16 As[32 * 40];  // 32 rows x 32 K-halfs, stride 40 (80B)
  const int tid = threadIdx.x;
  const int lane = tid & 31;
  const int wave = tid >> 5;
  const long m0 = (long)blockIdx.x * 32;
  const int wn = blockIdx.y * 128 + wave * 32;

  v8f acc[2][2] = {};
  for (int k0 = 0; k0 < K; k0 += 32) {
    {  // async stage A tile: 128 threads x 16B
      const int row = tid & 31, chunk = tid >> 5;
      cp_async_b128(X + (m0 + row) * (long)ldx + k0 + chunk * 8,
                    As + row * 40 + chunk * 8);
      wait_async();
    }
    __syncthreads();
    v16h af0 = lds_a_frag(As, 40, 0);
    v16h af1 = lds_a_frag(As + 16 * 40, 40, 0);
    v16h bf0 = glb_b_frag(W, ldw, wn, k0);
    v16h bf1 = glb_b_frag(W, ldw, wn + 16, k0);
    acc[0][0] = wmma16(af0, bf0, acc[0][0]);
    acc[0][1] = wmma16(af0, bf1, acc[0][1]);
    acc[1][0] = wmma16(af1, bf0, acc[1][0]);
    acc[1][1] = wmma16(af1, bf1, acc[1][1]);
    __syncthreads();
  }
#pragma unroll
  for (int mt = 0; mt < 2; ++mt)
#pragma unroll
    for (int nt = 0; nt < 2; ++nt)
#pragma unroll
      for (int r = 0; r < 8; ++r) {
        const long row = m0 + mt * 16 + r + 8 * (lane >> 4);
        const int cw = wn + nt * 16 + (lane & 15);
        float v = acc[mt][nt][r] + bias[cw];
        if (RESID) v += resid[row * 128 + cw];
        if (OUTH)
          Yh[row * (long)ldy + col_base + cw] = (_Float16)v;
        else
          Yf[row * (long)ldy + col_base + cw] = v;
      }
}

// ---------------------------------------------------------------------------
// Fused FFN: Y = relu(X @ W1^T + b1) @ W2^T + b2 + Xres (per 16-row block).
// ---------------------------------------------------------------------------
#define FFN_A_STRIDE 136   // halfs (272B, 16B-aligned rows)
#define FFN_H_STRIDE 2056  // halfs
#define FFN_LDS_BYTES ((16 * FFN_A_STRIDE + 16 * FFN_H_STRIDE) * 2)

__global__ __launch_bounds__(128) void ffn_kernel(
    const _Float16* __restrict__ Xh, const float* __restrict__ Xres,
    const _Float16* __restrict__ W1, const float* __restrict__ b1,
    const _Float16* __restrict__ W2, const float* __restrict__ b2,
    float* __restrict__ Y) {
  extern __shared__ _Float16 sm[];
  _Float16* Asm = sm;                      // [16][136]
  _Float16* Hsm = sm + 16 * FFN_A_STRIDE;  // [16][2056]
  const int tid = threadIdx.x;
  const int lane = tid & 31;
  const int wave = tid >> 5;
  const long m0 = (long)blockIdx.x * 16;

  {  // async stage 16x128 f16 A tile: 128 threads x 2 x 16B
    const int row = tid >> 3, seg = tid & 7;
    const _Float16* g = Xh + (m0 + row) * 128 + seg * 16;
    _Float16* l = Asm + row * FFN_A_STRIDE + seg * 16;
    cp_async_b128(g, l);
    cp_async_b128(g + 8, l + 8);
    wait_async();
  }
  __syncthreads();

  for (int j = 0; j < 32; ++j) {  // phase 1: relu(A @ W1^T + b1) -> LDS f16
    const int n0 = wave * 512 + j * 16;
    v8f acc = {};
#pragma unroll
    for (int ks = 0; ks < 4; ++ks) {
      v16h af = lds_a_frag(Asm, FFN_A_STRIDE, ks * 32);
      v16h bf = glb_b_frag(W1, 128, n0, ks * 32);
      acc = wmma16(af, bf, acc);
    }
    const int col = n0 + (lane & 15);
    const float bc = b1[col];
#pragma unroll
    for (int r = 0; r < 8; ++r) {
      const int rl = r + 8 * (lane >> 4);
      Hsm[rl * FFN_H_STRIDE + col] = (_Float16)fmaxf(acc[r] + bc, 0.f);
    }
  }
  __syncthreads();

  v8f acc2[2] = {};  // phase 2: H @ W2^T + b2 + Xres
  for (int ks = 0; ks < 64; ++ks) {
    v16h af = lds_a_frag(Hsm, FFN_H_STRIDE, ks * 32);
#pragma unroll
    for (int nt = 0; nt < 2; ++nt) {
      v16h bf = glb_b_frag(W2, 2048, wave * 32 + nt * 16, ks * 32);
      acc2[nt] = wmma16(af, bf, acc2[nt]);
    }
  }
#pragma unroll
  for (int nt = 0; nt < 2; ++nt)
#pragma unroll
    for (int r = 0; r < 8; ++r) {
      const long row = m0 + r + 8 * (lane >> 4);
      const int col = wave * 32 + nt * 16 + (lane & 15);
      Y[row * 128 + col] = acc2[nt][r] + b2[col] + Xres[row * 128 + col];
    }
}

// ---------------------------------------------------------------------------
// Flash attention, one wave per (16-query tile, batch, head). dh = 16.
// ---------------------------------------------------------------------------
template <int S, bool MASKED>
__global__ __launch_bounds__(32) void attn_kernel(
    const _Float16* __restrict__ qkv, _Float16* __restrict__ out,
    int seq_stride, int batch_stride) {
  __shared__ _Float16 P[16 * 34];
  const int lane = threadIdx.x & 31;
  const int half = lane >> 4, lo = lane & 15;
  const int q0 = blockIdx.x * 16;
  const int b = blockIdx.y;
  const int h = blockIdx.z;

  v16h qf;  // Q fragment x 1/sqrt(dh); contraction padded 16->32
  {
    const long tok = (long)(q0 + lo) * seq_stride + (long)b * batch_stride;
    const _Float16* p = qkv + tok * 384 + h * 16;
#pragma unroll
    for (int i = 0; i < 16; ++i)
      qf[i] = (half == 0) ? (_Float16)((float)p[i] * 0.25f) : (_Float16)0.f;
  }

  float mr[8], lr[8];
#pragma unroll
  for (int r = 0; r < 8; ++r) {
    mr[r] = -1e30f;
    lr[r] = 0.f;
  }
  v8f o = {};

  for (int kc = 0; kc < S; kc += 32) {
    v16h kf0, kf1;  // B: lane col = key, contiguous K = 16 dh values
    {
      const long t0 = (long)(kc + lo) * seq_stride + (long)b * batch_stride;
      const long t1 = (long)(kc + 16 + lo) * seq_stride + (long)b * batch_stride;
      const _Float16* p0 = qkv + t0 * 384 + 128 + h * 16;
      const _Float16* p1 = qkv + t1 * 384 + 128 + h * 16;
#pragma unroll
      for (int i = 0; i < 16; ++i) {
        kf0[i] = (half == 0) ? p0[i] : (_Float16)0.f;  // zero padded K half
        kf1[i] = (half == 0) ? p1[i] : (_Float16)0.f;
      }
    }
    v8f zero = {};
    v8f s0 = wmma16(qf, kf0, zero);
    v8f s1 = wmma16(qf, kf1, zero);

    float p0v[8], p1v[8], alpha[8];
#pragma unroll
    for (int r = 0; r < 8; ++r) {
      float a0 = s0[r], a1 = s1[r];
      if (MASKED) {
        const int row = q0 + r + 8 * half;
        const int c0 = kc + lo, c1 = c0 + 16;
        if (!(c0 <= row && c0 > row - WNDW)) a0 = -1e30f;
        if (!(c1 <= row && c1 > row - WNDW)) a1 = -1e30f;
      }
      float cm = fmaxf(a0, a1);
#pragma unroll
      for (int m = 8; m >= 1; m >>= 1) cm = fmaxf(cm, __shfl_xor(cm, m, 32));
      const float mn = fmaxf(mr[r], cm);
      const float al = __expf(mr[r] - mn);
      const float e0 = __expf(a0 - mn);
      const float e1 = __expf(a1 - mn);
      float rs = e0 + e1;
#pragma unroll
      for (int m = 8; m >= 1; m >>= 1) rs += __shfl_xor(rs, m, 32);
      lr[r] = lr[r] * al + rs;
      mr[r] = mn;
      alpha[r] = al;
      p0v[r] = e0;
      p1v[r] = e1;
    }

#pragma unroll
    for (int r = 0; r < 8; ++r) {  // C-layout -> A-layout via LDS
      const int rl = r + 8 * half;
      P[rl * 34 + lo] = (_Float16)p0v[r];
      P[rl * 34 + 16 + lo] = (_Float16)p1v[r];
    }
    v16h pf = lds_a_frag(P, 34, 0);
    v16h vf;  // B[k=key][n=dim]: lane dim = lo, keys kc + 16*half + i
#pragma unroll
    for (int i = 0; i < 16; ++i) {
      const long tokv =
          (long)(kc + 16 * half + i) * seq_stride + (long)b * batch_stride;
      vf[i] = qkv[tokv * 384 + 256 + h * 16 + lo];
    }
#pragma unroll
    for (int r = 0; r < 8; ++r) o[r] *= alpha[r];
    o = wmma16(pf, vf, o);
  }

#pragma unroll
  for (int r = 0; r < 8; ++r) {
    const long tok =
        (long)(q0 + r + 8 * half) * seq_stride + (long)b * batch_stride;
    out[tok * 128 + h * 16 + lo] = (_Float16)(o[r] / lr[r]);
  }
}

// ---------------------------------------------------------------------------
__global__ __launch_bounds__(128) void ln_kernel(const float* __restrict__ X,
                                                 const float* __restrict__ g,
                                                 const float* __restrict__ b,
                                                 float* __restrict__ Y,
                                                 _Float16* __restrict__ Yh) {
  const int wave = threadIdx.x >> 5, lane = threadIdx.x & 31;
  const long row = (long)blockIdx.x * 4 + wave;
  const float4 v = *(const float4*)(X + row * 128 + lane * 4);
  float s = v.x + v.y + v.z + v.w;
#pragma unroll
  for (int m = 16; m >= 1; m >>= 1) s += __shfl_xor(s, m, 32);
  const float mean = s * (1.f / 128.f);
  const float dx = v.x - mean, dy = v.y - mean, dz = v.z - mean,
              dw = v.w - mean;
  float q = dx * dx + dy * dy + dz * dz + dw * dw;
#pragma unroll
  for (int m = 16; m >= 1; m >>= 1) q += __shfl_xor(q, m, 32);
  const float rinv = rsqrtf(q * (1.f / 128.f) + 1e-5f);
  const int c = lane * 4;
  float4 ov;
  ov.x = dx * rinv * g[c + 0] + b[c + 0];
  ov.y = dy * rinv * g[c + 1] + b[c + 1];
  ov.z = dz * rinv * g[c + 2] + b[c + 2];
  ov.w = dw * rinv * g[c + 3] + b[c + 3];
  *(float4*)(Y + row * 128 + c) = ov;
  Yh[row * 128 + c + 0] = (_Float16)ov.x;
  Yh[row * 128 + c + 1] = (_Float16)ov.y;
  Yh[row * 128 + c + 2] = (_Float16)ov.z;
  Yh[row * 128 + c + 3] = (_Float16)ov.w;
}

__global__ __launch_bounds__(256) void t2v_kernel(
    const float* __restrict__ x, const float* __restrict__ w,
    const float* __restrict__ bb, const float* __restrict__ w0,
    const float* __restrict__ b0, float* __restrict__ out,
    _Float16* __restrict__ outh) {
  const int idx = blockIdx.x * 256 + threadIdx.x;  // over MTOK*128
  const int d = idx & 127;
  const int tok = idx >> 7;
  const int t = tok >> 9;  // token = t*512 + n
  const float tau = (float)(t + 1);
  const float v = (d < 127) ? __sinf(tau * w[d] + bb[d]) : (tau * w0[0] + b0[0]);
  const float r = x[idx] + v;
  out[idx] = r;
  outh[idx] = (_Float16)r;
}

__global__ __launch_bounds__(256) void cvt_kernel(const float* __restrict__ s,
                                                  _Float16* __restrict__ d,
                                                  int n) {
  const int i = blockIdx.x * 256 + threadIdx.x;
  if (i < n) d[i] = (_Float16)s[i];
}

__global__ __launch_bounds__(128) void head_kernel(const float* __restrict__ X,
                                                   const float* __restrict__ w,
                                                   float* __restrict__ out) {
  const int wave = threadIdx.x >> 5, lane = threadIdx.x & 31;
  const long row = (long)blockIdx.x * 4 + wave;
  const float* p = X + row * 128 + lane * 4;
  const float* q = w + lane * 4;
  float s = p[0] * q[0] + p[1] * q[1] + p[2] * q[2] + p[3] * q[3];
#pragma unroll
  for (int m = 16; m >= 1; m >>= 1) s += __shfl_xor(s, m, 32);
  if (lane == 0) out[row] = s;
}

// ---------------------------------------------------------------------------
extern "C" void kernel_launch(void* const* d_in, const int* in_sizes, int n_in,
                              void* d_out, int out_size, void* d_ws,
                              size_t ws_size, hipStream_t stream) {
  const float* x = (const float*)d_in[0];
  const float* eiw = (const float*)d_in[1];
  const float* eib = (const float*)d_in[2];
  const float* eow = (const float*)d_in[3];
  const float* eob = (const float*)d_in[4];
  const float* el1w = (const float*)d_in[5];
  const float* el1b = (const float*)d_in[6];
  const float* el2w = (const float*)d_in[7];
  const float* el2b = (const float*)d_in[8];
  const float* eln1w = (const float*)d_in[9];
  const float* eln1b = (const float*)d_in[10];
  const float* eln2w = (const float*)d_in[11];
  const float* eln2b = (const float*)d_in[12];
  const float* qiw = (const float*)d_in[13];
  const float* qib = (const float*)d_in[14];
  const float* qow = (const float*)d_in[15];
  const float* qob = (const float*)d_in[16];
  const float* ql1w = (const float*)d_in[17];
  const float* ql1b = (const float*)d_in[18];
  const float* ql2w = (const float*)d_in[19];
  const float* ql2b = (const float*)d_in[20];
  const float* qln1w = (const float*)d_in[21];
  const float* qln1b = (const float*)d_in[22];
  const float* qln2w = (const float*)d_in[23];
  const float* qln2b = (const float*)d_in[24];
  const float* t2vw = (const float*)d_in[25];
  const float* t2vb = (const float*)d_in[26];
  const float* t2vw0 = (const float*)d_in[27];
  const float* t2vb0 = (const float*)d_in[28];
  const float* outw = (const float*)d_in[29];

  float* ws = (float*)d_ws;
  float* t1 = ws;
  float* t2 = t1 + (size_t)MTOK * 128;
  float* zs = t2 + (size_t)MTOK * 128;
  float* cur = zs + (size_t)MTOK * 128;

  _Float16* hb = (_Float16*)(cur + (size_t)MTOK * 128);
  _Float16* qkvh = hb;
  _Float16* xh = qkvh + (size_t)MTOK * 384;
  _Float16* atth = xh + (size_t)MTOK * 128;
  _Float16* t2h = atth + (size_t)MTOK * 128;
  _Float16* zsh = t2h + (size_t)MTOK * 128;
  _Float16* curh = zsh + (size_t)MTOK * 128;
  _Float16* w_eiw = curh + (size_t)MTOK * 128;
  _Float16* w_eow = w_eiw + 384 * 128;
  _Float16* w_el1 = w_eow + 128 * 128;
  _Float16* w_el2 = w_el1 + 2048 * 128;
  _Float16* w_qiw = w_el2 + 128 * 2048;
  _Float16* w_qow = w_qiw + 2 * 384 * 128;
  _Float16* w_ql1 = w_qow + 2 * 128 * 128;
  _Float16* w_ql2 = w_ql1 + 2 * 2048 * 128;

  const dim3 blk(128);
  auto cvt = [&](const float* s, _Float16* d, int n) {
    cvt_kernel<<<dim3((n + 255) / 256), dim3(256), 0, stream>>>(s, d, n);
  };

  cvt(x, xh, MTOK * 128);
  cvt(eiw, w_eiw, 384 * 128);
  cvt(eow, w_eow, 128 * 128);
  cvt(el1w, w_el1, 2048 * 128);
  cvt(el2w, w_el2, 128 * 2048);
  cvt(qiw, w_qiw, 2 * 384 * 128);
  cvt(qow, w_qow, 2 * 128 * 128);
  cvt(ql1w, w_ql1, 2 * 2048 * 128);
  cvt(ql2w, w_ql2, 2 * 128 * 2048);

  gemm_kernel<false, true><<<dim3(MTOK / 32, 3), blk, 0, stream>>>(
      xh, 128, w_eiw, 128, eib, nullptr, nullptr, qkvh, 384, 0, 128);
  attn_kernel<NN, false><<<dim3(NN / 16, TT, HH), dim3(32), 0, stream>>>(
      qkvh, atth, 1, NN);
  gemm_kernel<true, false><<<dim3(MTOK / 32, 1), blk, 0, stream>>>(
      atth, 128, w_eow, 128, eob, x, t1, nullptr, 128, 0, 128);
  ln_kernel<<<dim3(MTOK / 4), blk, 0, stream>>>(t1, eln1w, eln1b, t2, t2h);
  ffn_kernel<<<dim3(MTOK / 16), blk, FFN_LDS_BYTES, stream>>>(
      t2h, t2, w_el1, el1b, w_el2, el2b, t1);
  ln_kernel<<<dim3(MTOK / 4), blk, 0, stream>>>(t1, eln2w, eln2b, zs, zsh);

  t2v_kernel<<<dim3(MTOK * 128 / 256), dim3(256), 0, stream>>>(
      x, t2vw, t2vb, t2vw0, t2vb0, cur, curh);

  for (int l = 0; l < 2; ++l) {
    gemm_kernel<false, true><<<dim3(MTOK / 32, 2), blk, 0, stream>>>(
        curh, 128, w_qiw + (size_t)l * 384 * 128, 128, qib + l * 384, nullptr,
        nullptr, qkvh, 384, 0, 128);
    gemm_kernel<false, true><<<dim3(MTOK / 32, 1), blk, 0, stream>>>(
        zsh, 128, w_qiw + (size_t)l * 384 * 128 + 256 * 128, 128,
        qib + l * 384 + 256, nullptr, nullptr, qkvh, 384, 256, 128);
    attn_kernel<TT, true><<<dim3(TT / 16, NN, HH), dim3(32), 0, stream>>>(
        qkvh, atth, NN, 1);
    gemm_kernel<true, false><<<dim3(MTOK / 32, 1), blk, 0, stream>>>(
        atth, 128, w_qow + (size_t)l * 128 * 128, 128, qob + l * 128, cur, t1,
        nullptr, 128, 0, 128);
    ln_kernel<<<dim3(MTOK / 4), blk, 0, stream>>>(t1, qln1w + l * 128,
                                                  qln1b + l * 128, t2, t2h);
    ffn_kernel<<<dim3(MTOK / 16), blk, FFN_LDS_BYTES, stream>>>(
        t2h, t2, w_ql1 + (size_t)l * 2048 * 128, ql1b + l * 2048,
        w_ql2 + (size_t)l * 128 * 2048, ql2b + l * 128, t1);
    ln_kernel<<<dim3(MTOK / 4), blk, 0, stream>>>(t1, qln2w + l * 128,
                                                  qln2b + l * 128, cur, curh);
  }

  head_kernel<<<dim3(MTOK / 4), blk, 0, stream>>>(cur, outw, (float*)d_out);
}
